// SelfAttention_40656160424626
// MI455X (gfx1250) — matrix operand order repaired and checked by reading, not compiled, not run on hardware
//
#include <hip/hip_runtime.h>

typedef __attribute__((ext_vector_type(16))) __bf16 v16bf;
typedef __attribute__((ext_vector_type(8)))  float  v8f;
typedef __attribute__((ext_vector_type(4)))  unsigned int v4u;
typedef __attribute__((ext_vector_type(8)))  int v8i;
typedef __attribute__((ext_vector_type(4)))  int v4i;

union BFrag {
    v16bf v;
    unsigned int u[8];
};
union BfPair {
    unsigned int u;
    __bf16 h[2];
};

#define DIM 768
#define NH 12
#define SLEN 2048
#define HD 64
#define NEG_BIG (-1e30f)

__device__ __forceinline__ v8f wmma_bf16(const BFrag& a, const BFrag& b, v8f c) {
    return __builtin_amdgcn_wmma_f32_16x16x32_bf16(false, a.v, false, b.v,
                                                   (short)0, c, false, false);
}

// A-operand K index for pair j (contiguous bf16 pair), 16x32 tile
__device__ __forceinline__ int kA(int lane, int j) {
    return ((lane >> 4) * 8) + 2 * j + ((j >= 4) ? 8 : 0);
}
// B-operand K index for pair j, 32x16 tile (same order as kA)
__device__ __forceinline__ int kB(int lane, int j) {
    return ((lane >> 4) * 8) + 2 * j + ((j >= 4) ? 8 : 0);
}

// ---------------------------------------------------------------------------
// TDM: issue a 2-D bf16 tile load global->LDS via the Tensor Data Mover.
// ---------------------------------------------------------------------------
__device__ __forceinline__ void tdm_load_2d_bf16(
    unsigned int lds_off, const void* gptr,
    unsigned int tile_d0, unsigned int tile_d1,
    unsigned int t_d0, unsigned int t_d1,
    unsigned int d0_stride, unsigned int padbits)
{
    unsigned long long ga = (unsigned long long)gptr;
    v4u g0;
    g0.x = 1u;                                   // count=1, user descriptor
    g0.y = lds_off;                              // lds_addr
    g0.z = (unsigned int)ga;                     // global_addr[31:0]
    g0.w = (unsigned int)((ga >> 32) & 0x01FFFFFFull) | (2u << 30);  // type=2
    v8i g1;
    g1[0] = (int)((1u << 16) | padbits);         // data_size=2B + pad config
    g1[1] = (int)(t_d0 << 16);                   // tensor_dim0[15:0] @ bit48
    g1[2] = (int)((t_d0 >> 16) | (t_d1 << 16));  // dim0 hi, tensor_dim1 lo
    g1[3] = (int)(((t_d1 >> 16) & 0xFFFFu) | (tile_d0 << 16));  // tile_dim0
    g1[4] = (int)(tile_d1 & 0xFFFFu);            // tile_dim1 (tile_dim2=0)
    g1[5] = (int)d0_stride;                      // tensor_dim0_stride lo32
    g1[6] = 0;
    g1[7] = 0;
    v4i z4 = {0, 0, 0, 0};
    v8i z8 = {0, 0, 0, 0, 0, 0, 0, 0};
    __builtin_amdgcn_tensor_load_to_lds(g0, g1, z4, z4, z8, 0);
}

// ---------------------------------------------------------------------------
// Kernel 1: fused QKV projection.  X[8192x768] fp32 -> Q,K,V bf16 [8192x768]
// Block 256 thr (8 waves, 4x2 tiles) computes 64 rows x 32 cols.
// All staging is pair-packed: float2 load -> v_cvt_pk_bf16_f32 -> b32 store.
// ---------------------------------------------------------------------------
__global__ __launch_bounds__(256) void qkv_proj_kernel(
    const float* __restrict__ X,
    const float* __restrict__ Wq, const float* __restrict__ Wk,
    const float* __restrict__ Wv,
    __bf16* __restrict__ Q, __bf16* __restrict__ K, __bf16* __restrict__ V)
{
    __shared__ __bf16 As[64][40];
    __shared__ __bf16 Bs[3][32][40];

    const int tid  = threadIdx.x;
    const int lane = tid & 31;
    const int wave = tid >> 5;
    const int wm   = wave & 3;   // row tile 0..3
    const int wn   = wave >> 2;  // col tile 0..1
    const int rowBase = blockIdx.x * 64;
    const int colBase = blockIdx.y * 32;

    v8f cq = {}; v8f ck = {}; v8f cv = {};

    for (int kt = 0; kt < DIM; kt += 32) {
        // A tile 64x32: 1024 bf16 pairs, 4 per thread
        #pragma unroll
        for (int i = 0; i < 4; ++i) {
            int p = tid + i * 256;
            int r = p >> 4, c = (p & 15) * 2;
            const float2 xv = *(const float2*)&X[(size_t)(rowBase + r) * DIM + kt + c];
            BfPair bp; bp.h[0] = (__bf16)xv.x; bp.h[1] = (__bf16)xv.y;
            *(unsigned int*)&As[r][c] = bp.u;
        }
        // W slices 32x32 each: 512 pairs, 2 per thread per matrix
        #pragma unroll
        for (int i = 0; i < 2; ++i) {
            int p = tid + i * 256;
            int r = p >> 4, c = (p & 15) * 2;
            size_t off = (size_t)(colBase + r) * DIM + kt + c;
            float2 q2 = *(const float2*)&Wq[off];
            float2 k2 = *(const float2*)&Wk[off];
            float2 v2 = *(const float2*)&Wv[off];
            BfPair bq2; bq2.h[0] = (__bf16)q2.x; bq2.h[1] = (__bf16)q2.y;
            BfPair bk2; bk2.h[0] = (__bf16)k2.x; bk2.h[1] = (__bf16)k2.y;
            BfPair bv2; bv2.h[0] = (__bf16)v2.x; bv2.h[1] = (__bf16)v2.y;
            *(unsigned int*)&Bs[0][r][c] = bq2.u;
            *(unsigned int*)&Bs[1][r][c] = bk2.u;
            *(unsigned int*)&Bs[2][r][c] = bv2.u;
        }
        __syncthreads();

        BFrag a, bq, bk, bv;
        const __bf16* arow = &As[wm * 16 + (lane & 15)][0];
        const __bf16* q0 = &Bs[0][wn * 16 + (lane & 15)][0];
        const __bf16* k0 = &Bs[1][wn * 16 + (lane & 15)][0];
        const __bf16* v0 = &Bs[2][wn * 16 + (lane & 15)][0];
        #pragma unroll
        for (int j = 0; j < 8; ++j) {
            a.u[j]  = *(const unsigned int*)(arow + kA(lane, j));
            int kb = kB(lane, j);
            bq.u[j] = *(const unsigned int*)(q0 + kb);
            bk.u[j] = *(const unsigned int*)(k0 + kb);
            bv.u[j] = *(const unsigned int*)(v0 + kb);
        }
        cq = wmma_bf16(a, bq, cq);
        ck = wmma_bf16(a, bk, ck);
        cv = wmma_bf16(a, bv, cv);
        __syncthreads();
    }

    const int N  = colBase + wn * 16 + (lane & 15);
    const int Mb = rowBase + wm * 16 + (lane >> 4) * 8;
    #pragma unroll
    for (int r = 0; r < 8; ++r) {
        size_t off = (size_t)(Mb + r) * DIM + N;
        Q[off] = (__bf16)cq[r];
        K[off] = (__bf16)ck[r];
        V[off] = (__bf16)cv[r];
    }
}

// ---------------------------------------------------------------------------
// Kernel 2: flash attention.  Q,K,V bf16 [B*H, S, 64] -> ctx fp32 [B*H, S, 64]
// Block 128 thr (4 waves); each wave owns 16 queries of one (b,h); the block
// streams shared 32-key K/V chunks with double buffering: while chunk i is
// consumed by WMMAs, the TDM fetches K(i+1) (wave 0 issues the D#) and all
// waves stage V(i+1) transposed.  LDS K row padding 64->72 via D# pad fields.
// ---------------------------------------------------------------------------
__global__ __launch_bounds__(128) void attn_kernel(
    const __bf16* __restrict__ Q,
    const __bf16* __restrict__ Kb,
    const __bf16* __restrict__ Vb,
    const int* __restrict__ mask,   // [B, S]
    float* __restrict__ ctx)
{
    __shared__ __bf16 Ks[2][32][72];     // keys x d (+8 halves pad per row)
    __shared__ __bf16 Vt[2][64][40];     // d x keys (transposed)
    __shared__ __bf16 Ps[4][16][40];     // per-wave P staging (C->A layout)

    const int tid  = threadIdx.x;
    const int lane = tid & 31;
    const int wave = tid >> 5;
    const int bh   = blockIdx.x;          // 0..B*H-1
    const int b    = bh / NH;
    const int qBase = blockIdx.y * 64 + wave * 16;
    const size_t base = (size_t)bh * SLEN * HD;
    const float qk_scale = 0.125f;        // 1/sqrt(64)
    // pad: enable | interval=32 DWORDs (code 4) | amount=4 DWORDs (code 3)
    const unsigned int padbits = (1u << 20) | (4u << 22) | (3u << 25);
    const unsigned int ksOff0 = (unsigned int)(size_t)&Ks[0][0][0];
    const unsigned int ksOff1 = (unsigned int)(size_t)&Ks[1][0][0];

    // stage a 32-key V chunk transposed, pair-packed along keys
    auto stageV = [&](int kbase, int buf) {
        #pragma unroll
        for (int i = 0; i < 8; ++i) {
            int p  = tid + i * 128;
            int r2 = (p >> 6) * 2;     // key pair 0..30
            int c  = p & 63;           // d, coalesced
            BfPair bp;
            bp.h[0] = Vb[base + (size_t)(kbase + r2) * HD + c];
            bp.h[1] = Vb[base + (size_t)(kbase + r2 + 1) * HD + c];
            *(unsigned int*)&Vt[buf][c][r2] = bp.u;
        }
    };

    // Q fragments for the two D=64 k-steps (A layout)
    BFrag qf[2];
    {
        const __bf16* qrow = Q + base + (size_t)(qBase + (lane & 15)) * HD;
        #pragma unroll
        for (int s = 0; s < 2; ++s)
            #pragma unroll
            for (int j = 0; j < 8; ++j)
                qf[s].u[j] = *(const unsigned int*)(qrow + s * 32 + kA(lane, j));
    }

    v8f o[4] = {{}, {}, {}, {}};
    float m[8], l[8];
    #pragma unroll
    for (int r = 0; r < 8; ++r) { m[r] = NEG_BIG; l[r] = 0.f; }

    const int NCHUNK = SLEN / 32;

    // pipeline prologue: chunk 0
    if (wave == 0)
        tdm_load_2d_bf16(ksOff0, Kb + base, HD, 32, HD, SLEN, HD, padbits);
    stageV(0, 0);
    if (wave == 0) __builtin_amdgcn_s_wait_tensorcnt(0);
    __syncthreads();

    for (int ci = 0; ci < NCHUNK; ++ci) {
        const int cur = ci & 1;
        const int nxt = cur ^ 1;
        const int kb0 = ci * 32;

        // prefetch chunk ci+1 while computing on chunk ci
        if (ci + 1 < NCHUNK) {
            if (wave == 0)
                tdm_load_2d_bf16(cur ? ksOff0 : ksOff1,
                                 Kb + base + (size_t)(kb0 + 32) * HD,
                                 HD, 32, HD, (unsigned)(SLEN - kb0 - 32),
                                 HD, padbits);
            stageV(kb0 + 32, nxt);
        }

        // scores: two 16-key column tiles, k-dim = 64 (2 WMMA steps each)
        v8f c0 = {}; v8f c1 = {};
        #pragma unroll
        for (int s = 0; s < 2; ++s) {
            BFrag kf0, kf1;
            const __bf16* kr0 = &Ks[cur][(lane & 15)][s * 32];
            const __bf16* kr1 = &Ks[cur][16 + (lane & 15)][s * 32];
            #pragma unroll
            for (int j = 0; j < 8; ++j) {
                int kb = kB(lane, j);
                kf0.u[j] = *(const unsigned int*)(kr0 + kb);
                kf1.u[j] = *(const unsigned int*)(kr1 + kb);
            }
            c0 = wmma_bf16(qf[s], kf0, c0);
            c1 = wmma_bf16(qf[s], kf1, c1);
        }

        // mask + scale (each lane owns one key column per tile)
        const int key0 = kb0 + (lane & 15);
        const bool ok0 = mask[b * SLEN + key0] != 0;
        const bool ok1 = mask[b * SLEN + key0 + 16] != 0;
        #pragma unroll
        for (int r = 0; r < 8; ++r) {
            c0[r] = ok0 ? c0[r] * qk_scale : NEG_BIG;
            c1[r] = ok1 ? c1[r] * qk_scale : NEG_BIG;
        }

        // online softmax: row stats via 16-lane xor-shuffle reductions
        float alpha[8], p0[8], p1[8];
        #pragma unroll
        for (int r = 0; r < 8; ++r) {
            float v = fmaxf(c0[r], c1[r]);
            #pragma unroll
            for (int d = 1; d < 16; d <<= 1)
                v = fmaxf(v, __shfl_xor(v, d, 32));
            float mn = fmaxf(m[r], v);
            float al = __expf(m[r] - mn);
            p0[r] = __expf(c0[r] - mn);
            p1[r] = __expf(c1[r] - mn);
            float rs = p0[r] + p1[r];
            #pragma unroll
            for (int d = 1; d < 16; d <<= 1)
                rs += __shfl_xor(rs, d, 32);
            l[r] = l[r] * al + rs;
            m[r] = mn;
            alpha[r] = al;
        }
        #pragma unroll
        for (int n = 0; n < 4; ++n)
            #pragma unroll
            for (int r = 0; r < 8; ++r)
                o[n][r] *= alpha[r];

        // P: C-layout -> LDS -> A-layout (same-wave DS ops are in-order)
        {
            int col = lane & 15;
            int rowOff = (lane >> 4) * 8;
            #pragma unroll
            for (int r = 0; r < 8; ++r) {
                Ps[wave][rowOff + r][col]      = (__bf16)p0[r];
                Ps[wave][rowOff + r][16 + col] = (__bf16)p1[r];
            }
        }
        BFrag pf;
        {
            const __bf16* prow = &Ps[wave][lane & 15][0];
            #pragma unroll
            for (int j = 0; j < 8; ++j)
                pf.u[j] = *(const unsigned int*)(prow + kA(lane, j));
        }

        // O += P(16x32) x V(32x64): 4 n-tiles
        #pragma unroll
        for (int n = 0; n < 4; ++n) {
            BFrag vf;
            const __bf16* vrow = &Vt[cur][n * 16 + (lane & 15)][0];
            #pragma unroll
            for (int j = 0; j < 8; ++j)
                vf.u[j] = *(const unsigned int*)(vrow + kB(lane, j));
            o[n] = wmma_bf16(pf, vf, o[n]);
        }

        if ((ci + 1 < NCHUNK) && wave == 0)
            __builtin_amdgcn_s_wait_tensorcnt(0);
        __syncthreads();
    }

    // normalize + store ctx
    float inv[8];
    #pragma unroll
    for (int r = 0; r < 8; ++r) inv[r] = (l[r] > 0.f) ? 1.f / l[r] : 0.f;
    const int M0  = qBase + (lane >> 4) * 8;
    const int col = lane & 15;
    #pragma unroll
    for (int n = 0; n < 4; ++n)
        #pragma unroll
        for (int r = 0; r < 8; ++r)
            ctx[base + (size_t)(M0 + r) * HD + n * 16 + col] = o[n][r] * inv[r];
}

// ---------------------------------------------------------------------------
// Kernel 3: output projection.  ctx fp32 [8192x768] @ wo^T -> out fp32
// ---------------------------------------------------------------------------
__global__ __launch_bounds__(256) void out_proj_kernel(
    const float* __restrict__ X,
    const float* __restrict__ W,
    float* __restrict__ Out)
{
    __shared__ __bf16 As[64][40];
    __shared__ __bf16 Bs[32][40];

    const int tid  = threadIdx.x;
    const int lane = tid & 31;
    const int wave = tid >> 5;
    const int wm   = wave & 3;
    const int wn   = wave >> 2;
    const int rowBase = blockIdx.x * 64;
    const int colBase = blockIdx.y * 32;

    v8f acc = {};

    for (int kt = 0; kt < DIM; kt += 32) {
        #pragma unroll
        for (int i = 0; i < 4; ++i) {
            int p = tid + i * 256;
            int r = p >> 4, c = (p & 15) * 2;
            const float2 xv = *(const float2*)&X[(size_t)(rowBase + r) * DIM + kt + c];
            BfPair bp; bp.h[0] = (__bf16)xv.x; bp.h[1] = (__bf16)xv.y;
            *(unsigned int*)&As[r][c] = bp.u;
        }
        #pragma unroll
        for (int i = 0; i < 2; ++i) {
            int p = tid + i * 256;
            int r = p >> 4, c = (p & 15) * 2;
            const float2 wv2 = *(const float2*)&W[(size_t)(colBase + r) * DIM + kt + c];
            BfPair bp; bp.h[0] = (__bf16)wv2.x; bp.h[1] = (__bf16)wv2.y;
            *(unsigned int*)&Bs[r][c] = bp.u;
        }
        __syncthreads();

        BFrag a, bw;
        const __bf16* arow = &As[wm * 16 + (lane & 15)][0];
        const __bf16* brow = &Bs[wn * 16 + (lane & 15)][0];
        #pragma unroll
        for (int j = 0; j < 8; ++j) {
            a.u[j]  = *(const unsigned int*)(arow + kA(lane, j));
            bw.u[j] = *(const unsigned int*)(brow + kB(lane, j));
        }
        acc = wmma_bf16(a, bw, acc);
        __syncthreads();
    }

    const int N  = colBase + wn * 16 + (lane & 15);
    const int Mb = rowBase + wm * 16 + (lane >> 4) * 8;
    #pragma unroll
    for (int r = 0; r < 8; ++r)
        Out[(size_t)(Mb + r) * DIM + N] = acc[r];
}

// ---------------------------------------------------------------------------
extern "C" void kernel_launch(void* const* d_in, const int* in_sizes, int n_in,
                              void* d_out, int out_size, void* d_ws, size_t ws_size,
                              hipStream_t stream) {
    const float* h    = (const float*)d_in[0];
    const int*   mask = (const int*)d_in[1];
    const float* wq   = (const float*)d_in[2];
    const float* wk   = (const float*)d_in[3];
    const float* wv   = (const float*)d_in[4];
    const float* wo   = (const float*)d_in[5];
    float* out = (float*)d_out;

    const size_t R = 4 * 2048;          // B*S
    const size_t E = R * DIM;           // elements per projection buffer
    __bf16* Q = (__bf16*)d_ws;
    __bf16* K = Q + E;
    __bf16* V = K + E;
    float* ctx = (float*)(V + E);

    dim3 gProj(128, 24);                // 8192/64 x 768/32
    qkv_proj_kernel<<<gProj, dim3(256), 0, stream>>>(h, wq, wk, wv, Q, K, V);

    dim3 gAttn(48, 32);                 // B*H x S/64
    attn_kernel<<<gAttn, dim3(128), 0, stream>>>(Q, K, V, mask, ctx);

    out_proj_kernel<<<gProj, dim3(256), 0, stream>>>(ctx, wo, out);
}
